// MultiHeadPhasorBlock_45337674776746
// MI455X (gfx1250) — matrix operand order repaired and checked by reading, not compiled, not run on hardware
//
#include <hip/hip_runtime.h>
#include <hip/hip_bf16.h>
#include <math.h>

// ---------------------------------------------------------------------------
// Types for CDNA5 WMMA
// ---------------------------------------------------------------------------
typedef __attribute__((ext_vector_type(16))) _Float16 v16h;
typedef __attribute__((ext_vector_type(8)))  _Float16 v8h;
typedef __attribute__((ext_vector_type(4)))  _Float16 v4h;
typedef __attribute__((ext_vector_type(8)))  float    v8f;

#define PI_F 3.14159265358979323846f

// Activation selectors for the GEMM epilogue
#define ACT_NONE   0
#define ACT_SIG    1
#define ACT_TANHPI 2
#define ACT_GELU   3

// ---------------------------------------------------------------------------
// Tiled WMMA GEMM:  C[row, colOff+col] = epi(A[M,K] @ W[K,N] + bias)
//   A : f32 row-major [M,K];  W : f32 row-major [K,N]
//   C : f32, row stride ldC, column offset colOff (write into comb slices)
//   f32 -> f16 conversion while staging tiles in LDS (vectorized b128/b64
//   DS stores); f32 WMMA accumulate; next K-tile globals preloaded into
//   registers while the current tile's WMMAs run.
//   BM=128 (8 waves x 16 rows), BN=64 or 32. K % 32 == 0, M % 128 == 0,
//   N % BN == 0 (true for every use here).
// ---------------------------------------------------------------------------
template <int ACT, int BN>
__global__ __launch_bounds__(256) void wmma_gemm_kernel(
    const float* __restrict__ A, const float* __restrict__ W,
    const float* __restrict__ bias, float* __restrict__ C,
    int M, int N, int K, int ldC, int colOff,
    const float* __restrict__ scale, const float* __restrict__ residual)
{
    constexpr int BM = 128;
    constexpr int NT = BN / 16;              // N-tiles per wave
    constexpr int KG = (BN == 64) ? 8 : 4;   // K's per B-staging thread
    __shared__ __align__(32) _Float16 Ash[BM * 32];   // [row][k]
    __shared__ __align__(32) _Float16 Bsh[BN * 32];   // [col][k] (transposed)

    const int tid  = threadIdx.x;
    const int lane = tid & 31;
    const int wv   = tid >> 5;               // wave id 0..7 -> M-tile
    const int rowBase = blockIdx.x * BM;
    const int colBase = blockIdx.y * BN;

    v8f zero = {};
    v8f acc[NT];
#pragma unroll
    for (int t = 0; t < NT; ++t) acc[t] = zero;

    // --- A staging: thread owns 16 contiguous f32 of half a row ---
    const int arow = tid >> 1;               // 0..127
    const int acol = (tid & 1) * 16;         // 0 or 16
    const float* Aptr = A + (size_t)(rowBase + arow) * K + acol;

    // --- B staging: thread owns column bn, K-group [bk, bk+KG) ---
    const int bn = tid % BN;
    const int bk = (tid / BN) * KG;
    const float* Wptr = W + (size_t)bk * N + colBase + bn;

    float4 areg[4];
    float  breg[KG];

    auto loadA = [&](int k0) {
        const float4* p = (const float4*)(Aptr + k0);
#pragma unroll
        for (int j = 0; j < 4; ++j) areg[j] = p[j];
    };
    auto loadB = [&](int k0) {
#pragma unroll
        for (int i = 0; i < KG; ++i) breg[i] = Wptr[(size_t)(k0 + i) * N];
    };
    auto storeA = [&]() {
        v8h lo, hi;
        lo[0] = (_Float16)areg[0].x; lo[1] = (_Float16)areg[0].y;
        lo[2] = (_Float16)areg[0].z; lo[3] = (_Float16)areg[0].w;
        lo[4] = (_Float16)areg[1].x; lo[5] = (_Float16)areg[1].y;
        lo[6] = (_Float16)areg[1].z; lo[7] = (_Float16)areg[1].w;
        hi[0] = (_Float16)areg[2].x; hi[1] = (_Float16)areg[2].y;
        hi[2] = (_Float16)areg[2].z; hi[3] = (_Float16)areg[2].w;
        hi[4] = (_Float16)areg[3].x; hi[5] = (_Float16)areg[3].y;
        hi[6] = (_Float16)areg[3].z; hi[7] = (_Float16)areg[3].w;
        *(v8h*)&Ash[arow * 32 + acol]     = lo;   // 16B-aligned
        *(v8h*)&Ash[arow * 32 + acol + 8] = hi;   // 16B-aligned
    };
    auto storeB = [&]() {
        if (BN == 64) {
            v8h b;
#pragma unroll
            for (int i = 0; i < 8; ++i) b[i] = (_Float16)breg[i];
            *(v8h*)&Bsh[bn * 32 + bk] = b;        // 16B-aligned
        } else {
            v4h b;
#pragma unroll
            for (int i = 0; i < 4; ++i) b[i] = (_Float16)breg[i];
            *(v4h*)&Bsh[bn * 32 + bk] = b;        // 8B-aligned
        }
    };

    loadA(0);
    loadB(0);
    for (int k0 = 0; k0 < K; k0 += 32) {
        storeA();
        storeB();
        __syncthreads();
        // prefetch next K-tile while WMMAs consume LDS
        if (k0 + 32 < K) { loadA(k0 + 32); loadB(k0 + 32); }

        // --- A fragment (16x32, contiguous k order, same as B below) ---
        // lane<16 : row=lane,    K in {0..15}
        // lane>=16: row=lane-16, K in {16..31}
        const int mrow = (wv << 4) + (lane & 15);
        const _Float16* ap = &Ash[mrow * 32 + ((lane & 16) ? 16 : 0)];
        v8h alo = *(const v8h*)(ap);
        v8h ahi = *(const v8h*)(ap + 8);
        v16h a;
#pragma unroll
        for (int i = 0; i < 8; ++i) { a[i] = alo[i]; a[8 + i] = ahi[i]; }

        // --- B fragments + WMMA (B 32x16: lane<16 col=lane K=0..15,
        //     lane>=16 col=lane-16 K=16..31) ---
#pragma unroll
        for (int nt = 0; nt < NT; ++nt) {
            const int col = (nt << 4) + (lane & 15);
            v16h b = *(const v16h*)&Bsh[col * 32 + ((lane & 16) ? 16 : 0)];
            acc[nt] = __builtin_amdgcn_wmma_f32_16x16x32_f16(
                false, a, false, b, (short)0, acc[nt], false, false);
        }
        __syncthreads();
    }

    // --- epilogue: bias + activation + |scale| + residual ---
    const float sc = scale ? fabsf(*scale) : 1.0f;
#pragma unroll
    for (int nt = 0; nt < NT; ++nt) {
        const int colN = colBase + (nt << 4) + (lane & 15);
        const float bv = bias ? bias[colN] : 0.0f;
#pragma unroll
        for (int r = 0; r < 8; ++r) {
            // C/D layout: VGPR r -> row r (lanes 0-15) / r+8 (lanes 16-31)
            const int row = rowBase + (wv << 4) + r + ((lane & 16) ? 8 : 0);
            float v = acc[nt][r] + bv;
            if (ACT == ACT_SIG)         v = 1.0f / (1.0f + __expf(-v));
            else if (ACT == ACT_TANHPI) v = tanhf(v) * PI_F;
            else if (ACT == ACT_GELU)   v = 0.5f * v * (1.0f + erff(v * 0.70710678118654752f));
            v *= sc;
            const size_t idx = (size_t)row * ldC + colOff + colN;
            if (residual) v += residual[idx];
            C[idx] = v;
        }
    }
}

// ---------------------------------------------------------------------------
// gate = sigmoid(x @ w_gate + b_gate), N=H=4 (too narrow for WMMA)
// block = 128 threads = 4 waves, one wave per head; grid = M rows
// ---------------------------------------------------------------------------
__global__ __launch_bounds__(128) void gate_kernel(
    const float* __restrict__ x, const float* __restrict__ w,
    const float* __restrict__ b, float* __restrict__ gate, int K, int H)
{
    const int row  = blockIdx.x;
    const int lane = threadIdx.x & 31;
    const int h    = threadIdx.x >> 5;
    float s = 0.0f;
    for (int k = lane; k < K; k += 32)
        s += x[(size_t)row * K + k] * w[(size_t)k * H + h];
    for (int off = 16; off > 0; off >>= 1) s += __shfl_down(s, off, 32);
    if (lane == 0)
        gate[(size_t)row * H + h] = 1.0f / (1.0f + __expf(-(s + b[h])));
}

// ---------------------------------------------------------------------------
// concat dst[M, Da+Db] = [a[M,Da], bsrc[row % bRows, Db]]
// (bRows = L broadcasts pos_phases over batch; bRows = M is plain concat)
// ---------------------------------------------------------------------------
__global__ void concat_kernel(float* __restrict__ dst,
                              const float* __restrict__ a,
                              const float* __restrict__ bsrc,
                              int M, int Da, int Db, int bRows)
{
    const int Dt = Da + Db;
    const size_t total = (size_t)M * Dt;
    for (size_t i = (size_t)blockIdx.x * blockDim.x + threadIdx.x; i < total;
         i += (size_t)gridDim.x * blockDim.x) {
        const int row = (int)(i / Dt), col = (int)(i % Dt);
        dst[i] = (col < Da) ? a[(size_t)row * Da + col]
                            : bsrc[(size_t)(row % bRows) * Db + (col - Da)];
    }
}

// ---------------------------------------------------------------------------
// Fused chunked scans over L for the phasor memory and ctx-average.
// One column (b,d) per thread; grid.y = NC chunks of length L/NC.
// pass1: per-chunk sums of (cos*wv, sin*wv, mag, x);  wv = mag*v1 (fused)
// pass2: add chunk prefix, re-walk chunk, emit mem_cos/mem_sin (normalized)
//        and ctx_avg.
// ---------------------------------------------------------------------------
__global__ void scan1_kernel(const float* __restrict__ x,
                             const float* __restrict__ pos,
                             const float* __restrict__ mag,
                             const float* __restrict__ v1,
                             float* __restrict__ cbuf, int L, int D)
{
    const int q = blockIdx.x * blockDim.x + threadIdx.x;  // (b,d) column
    const int c = blockIdx.y;
    const int NC = gridDim.y, CLEN = L / NC;
    const int b = q / D, d = q % D;
    float sc = 0.f, ss = 0.f, sm = 0.f, sx = 0.f;
    for (int l = c * CLEN; l < (c + 1) * CLEN; ++l) {
        const size_t idx = ((size_t)b * L + l) * D + d;
        const float m  = mag[idx];
        const float wv = m * v1[idx];
        const float ph = pos[(size_t)l * D + d];
        sc += __cosf(ph) * wv;
        ss += __sinf(ph) * wv;
        sm += m;
        sx += x[idx];
    }
    float* o = cbuf + ((size_t)q * NC + c) * 4;
    o[0] = sc; o[1] = ss; o[2] = sm; o[3] = sx;
}

__global__ void scan2_kernel(const float* __restrict__ x,
                             const float* __restrict__ pos,
                             const float* __restrict__ mag,
                             const float* __restrict__ v1,
                             const float* __restrict__ cbuf,
                             float* __restrict__ memc, float* __restrict__ mems,
                             float* __restrict__ ctx, int L, int D)
{
    const int q = blockIdx.x * blockDim.x + threadIdx.x;
    const int c = blockIdx.y;
    const int NC = gridDim.y, CLEN = L / NC;
    const int b = q / D, d = q % D;
    float ac = 0.f, as = 0.f, am = 0.f, ax = 0.f;
    for (int cc = 0; cc < c; ++cc) {
        const float* o = cbuf + ((size_t)q * NC + cc) * 4;
        ac += o[0]; as += o[1]; am += o[2]; ax += o[3];
    }
    for (int l = c * CLEN; l < (c + 1) * CLEN; ++l) {
        const size_t idx = ((size_t)b * L + l) * D + d;
        const float m  = mag[idx];
        const float wv = m * v1[idx];
        const float ph = pos[(size_t)l * D + d];
        ac += __cosf(ph) * wv;
        as += __sinf(ph) * wv;
        am += m;
        ax += x[idx];
        const float inv_s = rsqrtf(am + 1e-8f);
        memc[idx] = ac * inv_s;
        mems[idx] = as * inv_s;
        ctx[idx]  = ax / (float)(l + 1);
    }
}

// ---------------------------------------------------------------------------
// Gate cumsum -> inv_nf = rsqrt(max(cumsum(gate),1)).  B*H=8 scans, 32 chunks
// each -> one 256-thread block does everything (LDS holds chunk sums).
// ---------------------------------------------------------------------------
__global__ __launch_bounds__(256) void gatescan_kernel(
    const float* __restrict__ gate, float* __restrict__ invnf,
    int L, int H)
{
    __shared__ float csum[256];
    const int tid  = threadIdx.x;
    const int pair = tid >> 5;   // (b,h) 0..7
    const int c    = tid & 31;   // chunk
    const int b = pair / H, h = pair % H;
    const int CLEN = L / 32;
    float s = 0.f;
    for (int l = c * CLEN; l < (c + 1) * CLEN; ++l)
        s += gate[((size_t)b * L + l) * H + h];
    csum[tid] = s;
    __syncthreads();
    float run = 0.f;
    for (int cc = 0; cc < c; ++cc) run += csum[(pair << 5) + cc];
    for (int l = c * CLEN; l < (c + 1) * CLEN; ++l) {
        run += gate[((size_t)b * L + l) * H + h];
        invnf[((size_t)b * L + l) * H + h] = rsqrtf(fmaxf(run, 1.0f));
    }
}

// ---------------------------------------------------------------------------
// Holographic KV binding scan. State (Ph x V = 256 cos + 256 sin) kept in
// registers, one element per thread. Chunked two-pass over L.
// grid = (NC, H, B), block = 256; thread t -> (p = t>>3, v = t&7)
// ---------------------------------------------------------------------------
__global__ __launch_bounds__(256) void kv1_kernel(
    const float* __restrict__ sph, const float* __restrict__ vals,
    const float* __restrict__ gate, float* __restrict__ kbuf,
    int L, int H, int Ph, int V)
{
    const int c = blockIdx.x, h = blockIdx.y, b = blockIdx.z;
    const int NC = gridDim.x, CLEN = L / NC;
    const int tid = threadIdx.x;
    const int p = tid >> 3, v = tid & 7;
    float sc = 0.f, ss = 0.f;
    for (int l = c * CLEN; l < (c + 1) * CLEN; ++l) {
        const size_t base = (size_t)b * L + l;
        const float g   = gate[base * H + h];
        const float s   = sph[base * (H * Ph) + h * Ph + p];
        const float val = vals[base * (H * V) + h * V + v] * g;
        sc += __cosf(s) * val;
        ss += __sinf(s) * val;
    }
    float* o = kbuf + ((((size_t)(b * H + h) * NC) + c) * 256 + tid) * 2;
    o[0] = sc; o[1] = ss;
}

__global__ __launch_bounds__(256) void kv2_kernel(
    const float* __restrict__ sph, const float* __restrict__ vals,
    const float* __restrict__ gate, const float* __restrict__ qph,
    const float* __restrict__ invnf, const float* __restrict__ kbuf,
    float* __restrict__ kvret, int L, int H, int Ph, int V)
{
    __shared__ float red[256];
    const int c = blockIdx.x, h = blockIdx.y, b = blockIdx.z;
    const int NC = gridDim.x, CLEN = L / NC;
    const int tid = threadIdx.x;
    const int p = tid >> 3, v = tid & 7;
    float rc = 0.f, rs = 0.f;
    for (int cc = 0; cc < c; ++cc) {
        const float* o = kbuf + ((((size_t)(b * H + h) * NC) + cc) * 256 + tid) * 2;
        rc += o[0]; rs += o[1];
    }
    const float invSqrtPh = rsqrtf((float)Ph);
    for (int l = c * CLEN; l < (c + 1) * CLEN; ++l) {
        const size_t base = (size_t)b * L + l;
        const float g   = gate[base * H + h];
        const float s   = sph[base * (H * Ph) + h * Ph + p];
        const float val = vals[base * (H * V) + h * V + v] * g;
        rc += __cosf(s) * val;
        rs += __sinf(s) * val;
        const float q = qph[base * (H * Ph) + h * Ph + p];
        red[tid] = __cosf(q) * rc + __sinf(q) * rs;
        __syncthreads();
        if (tid < V) {
            float sum = 0.f;
            for (int pp = 0; pp < Ph; ++pp) sum += red[tid + pp * V];
            kvret[base * (H * V) + h * V + tid] =
                sum * invnf[base * H + h] * invSqrtPh;
        }
        __syncthreads();
    }
}

// ---------------------------------------------------------------------------
// Phasor retrieval:
//   qc = cos(pos)cos(off) - sin(pos)sin(off);  qs = sin(pos)cos(off)+cos*sin
//   pos_ret = (mem_cos*qc + mem_sin*qs) / sqrt(D)
// ---------------------------------------------------------------------------
__global__ void phasor_kernel(const float* __restrict__ pos,
                              const float* __restrict__ offset,
                              const float* __restrict__ memc,
                              const float* __restrict__ mems,
                              float* __restrict__ posret,
                              int L, int D, int B)
{
    const size_t LD = (size_t)L * D;
    const size_t total = (size_t)B * LD;
    const float invSqrtD = rsqrtf((float)D);
    for (size_t i = (size_t)blockIdx.x * blockDim.x + threadIdx.x; i < total;
         i += (size_t)gridDim.x * blockDim.x) {
        const float ph = pos[i % LD];
        const float pc = __cosf(ph), ps = __sinf(ph);
        const float of = offset[i];
        const float oc = __cosf(of), os = __sinf(of);
        const float qc = pc * oc - ps * os;
        const float qs = ps * oc + pc * os;
        posret[i] = (memc[i] * qc + mems[i] * qs) * invSqrtD;
    }
}

// ---------------------------------------------------------------------------
// traj = x * mean(cos(query_phase), over H*Ph) written into comb[:,colOff:]
// block = 128 threads (= P), grid = M rows
// ---------------------------------------------------------------------------
__global__ __launch_bounds__(128) void traj_kernel(
    const float* __restrict__ qph, const float* __restrict__ x,
    float* __restrict__ comb, int P, int D, int ldC, int colOff)
{
    __shared__ float red[128];
    const int row = blockIdx.x, tid = threadIdx.x;
    red[tid] = __cosf(qph[(size_t)row * P + tid]);
    __syncthreads();
    for (int s = 64; s > 0; s >>= 1) {
        if (tid < s) red[tid] += red[tid + s];
        __syncthreads();
    }
    const float mean = red[0] / (float)P;
    for (int j = tid; j < D; j += 128)
        comb[(size_t)row * ldC + colOff + j] = x[(size_t)row * D + j] * mean;
}

// ---------------------------------------------------------------------------
// Row LayerNorm over Dt=1536 with affine params
// ---------------------------------------------------------------------------
__global__ __launch_bounds__(256) void ln_kernel(
    const float* __restrict__ comb, const float* __restrict__ g,
    const float* __restrict__ bb, float* __restrict__ out, int Dt)
{
    __shared__ float r1[256], r2[256];
    const int row = blockIdx.x, tid = threadIdx.x;
    float s1 = 0.f, s2 = 0.f;
    for (int j = tid; j < Dt; j += 256) {
        const float v = comb[(size_t)row * Dt + j];
        s1 += v; s2 += v * v;
    }
    r1[tid] = s1; r2[tid] = s2;
    __syncthreads();
    for (int s = 128; s > 0; s >>= 1) {
        if (tid < s) { r1[tid] += r1[tid + s]; r2[tid] += r2[tid + s]; }
        __syncthreads();
    }
    const float mean = r1[0] / (float)Dt;
    const float var  = r2[0] / (float)Dt - mean * mean;
    const float rstd = rsqrtf(var + 1e-5f);
    for (int j = tid; j < Dt; j += 256)
        out[(size_t)row * Dt + j] =
            (comb[(size_t)row * Dt + j] - mean) * rstd * g[j] + bb[j];
}

// ---------------------------------------------------------------------------
// Host-side GEMM launcher
// ---------------------------------------------------------------------------
template <int ACT, int BN>
static void launch_gemm(hipStream_t s, const float* A, const float* W,
                        const float* bias, float* C, int M, int N, int K,
                        int ldC, int colOff, const float* scale,
                        const float* residual)
{
    dim3 grid(M / 128, N / BN), block(256);
    wmma_gemm_kernel<ACT, BN><<<grid, block, 0, s>>>(
        A, W, bias, C, M, N, K, ldC, colOff, scale, residual);
}

// ---------------------------------------------------------------------------
extern "C" void kernel_launch(void* const* d_in, const int* in_sizes, int n_in,
                              void* d_out, int out_size, void* d_ws, size_t ws_size,
                              hipStream_t stream)
{
    (void)in_sizes; (void)n_in; (void)out_size; (void)ws_size;
    // Shapes
    const int B = 2, L = 4096, D = 512, P = 128, V = 8, H = 4;
    const int M = B * L;            // 8192
    const int Ph = P / H;           // 32
    const int HV = H * V;           // 32
    const int D3 = 3 * D;           // 1536

    // Inputs (setup_inputs order)
    const float* x       = (const float*)d_in[0];
    const float* pos     = (const float*)d_in[1];
    const float* w_v1    = (const float*)d_in[2];
    const float* b_v1    = (const float*)d_in[3];
    const float* w_o1    = (const float*)d_in[4];
    const float* b_o1    = (const float*)d_in[5];
    const float* w_off   = (const float*)d_in[6];
    const float* b_off   = (const float*)d_in[7];
    const float* w_mag   = (const float*)d_in[8];
    const float* b_mag   = (const float*)d_in[9];
    const float* magsc   = (const float*)d_in[10];
    const float* w_key   = (const float*)d_in[11];
    const float* b_key   = (const float*)d_in[12];
    const float* w_val   = (const float*)d_in[13];
    const float* b_val   = (const float*)d_in[14];
    const float* w_sk1   = (const float*)d_in[15];
    const float* b_sk1   = (const float*)d_in[16];
    const float* w_sk2   = (const float*)d_in[17];
    const float* b_sk2   = (const float*)d_in[18];
    const float* w_gate  = (const float*)d_in[19];
    const float* b_gate  = (const float*)d_in[20];
    const float* w_kv    = (const float*)d_in[21];
    const float* b_kv    = (const float*)d_in[22];
    const float* ln_g    = (const float*)d_in[23];
    const float* ln_b    = (const float*)d_in[24];
    const float* w_f1    = (const float*)d_in[25];
    const float* b_f1    = (const float*)d_in[26];
    const float* w_f2    = (const float*)d_in[27];
    const float* b_f2    = (const float*)d_in[28];
    float* out = (float*)d_out;

    // Workspace layout (bytes). Lifetimes allow aliasing noted below.
    const size_t MBy = 1ull << 20;
    char* w = (char*)d_ws;
    float* v1    = (float*)(w +   0 * MBy);  // 16MB; later: pos_ret
    float* mag   = (float*)(w +  16 * MBy);  // 16MB
    float* qph   = (float*)(w +  32 * MBy);  //  4MB  query phases
    float* vals  = (float*)(w +  36 * MBy);  //  1MB
    float* gateb = (float*)(w +  37 * MBy);  //  128KB
    float* invnf = (float*)(w +  38 * MBy);  //  128KB
    float* cbuf  = (float*)(w +  39 * MBy);  //  512KB scan chunk sums
    float* kbuf  = (float*)(w +  40 * MBy);  //  512KB kv chunk sums
    float* ctx   = (float*)(w +  41 * MBy);  // 16MB } these three (48MB)
    float* memc  = (float*)(w +  57 * MBy);  // 16MB }  later reused as lnbuf
    float* mems  = (float*)(w +  73 * MBy);  // 16MB }
    float* skin  = (float*)(w +  89 * MBy);  // 32MB; later: off_in, then h2
    float* hmid  = (float*)(w + 121 * MBy);  // 16MB; later: offset
    float* sph   = (float*)(w + 137 * MBy);  //  4MB  storage phases
    float* kvret = (float*)(w + 141 * MBy);  //  1MB
    float* comb  = (float*)(w + 142 * MBy);  // 48MB  -> total 190MB
    float* lnbuf = ctx;                      // aliases ctx/memc/mems region
    float* offin = skin;                     // alias
    float* offset = hmid;                    // alias
    float* posret = v1;                      // alias
    float* h2    = skin;                     // alias

    const int NC = 32;                       // scan chunks
    dim3 b256(256), b128(128);

    // ---- 1) independent projections of x (WMMA GEMMs) ----
    launch_gemm<ACT_NONE,  64>(stream, x, w_v1,  b_v1,  v1,   M, D,  D, D,  0, nullptr, nullptr);
    launch_gemm<ACT_SIG,   64>(stream, x, w_mag, b_mag, mag,  M, D,  D, D,  0, magsc,   nullptr);
    launch_gemm<ACT_TANHPI,64>(stream, x, w_key, b_key, qph,  M, P,  D, P,  0, nullptr, nullptr);
    launch_gemm<ACT_NONE,  32>(stream, x, w_val, b_val, vals, M, HV, D, HV, 0, nullptr, nullptr);
    gate_kernel<<<dim3(M), b128, 0, stream>>>(x, w_gate, b_gate, gateb, D, H);

    // ---- 2) chunked scans: phasor memory + ctx average ----
    scan1_kernel<<<dim3(B * D / 256, NC), b256, 0, stream>>>(x, pos, mag, v1, cbuf, L, D);
    scan2_kernel<<<dim3(B * D / 256, NC), b256, 0, stream>>>(x, pos, mag, v1, cbuf,
                                                             memc, mems, ctx, L, D);
    // ---- 3) gate cumsum -> inv_nf ----
    gatescan_kernel<<<dim3(1), b256, 0, stream>>>(gateb, invnf, L, H);

    // ---- 4) storage-key path: [x, ctx] -> gelu -> tanh*pi ----
    concat_kernel<<<dim3(2048), b256, 0, stream>>>(skin, x, ctx, M, D, D, M);
    launch_gemm<ACT_GELU,  64>(stream, skin, w_sk1, b_sk1, hmid, M, D, 2 * D, D, 0, nullptr, nullptr);
    launch_gemm<ACT_TANHPI,64>(stream, hmid, w_sk2, b_sk2, sph,  M, P, D,     P, 0, nullptr, nullptr);

    // ---- 5) KV binding scan + retrieval ----
    kv1_kernel<<<dim3(NC, H, B), b256, 0, stream>>>(sph, vals, gateb, kbuf, L, H, Ph, V);
    kv2_kernel<<<dim3(NC, H, B), b256, 0, stream>>>(sph, vals, gateb, qph, invnf,
                                                    kbuf, kvret, L, H, Ph, V);

    // ---- 6) offset path: [x, pos(bcast)] -> tanh*pi -> phasor retrieval ----
    concat_kernel<<<dim3(2048), b256, 0, stream>>>(offin, x, pos, M, D, D, L);
    launch_gemm<ACT_TANHPI,64>(stream, offin, w_off, b_off, offset, M, D, 2 * D, D, 0, nullptr, nullptr);
    phasor_kernel<<<dim3(4096), b256, 0, stream>>>(pos, offset, memc, mems, posret, L, D, B);

    // ---- 7) heads of comb = [pos_out | kv_out | traj] (GEMMs write in place) ----
    launch_gemm<ACT_NONE,64>(stream, posret, w_o1, b_o1, comb, M, D, D,  D3, 0,     nullptr, nullptr);
    launch_gemm<ACT_NONE,64>(stream, kvret,  w_kv, b_kv, comb, M, D, HV, D3, D,     nullptr, nullptr);
    traj_kernel<<<dim3(M), b128, 0, stream>>>(qph, x, comb, P, D, D3, 2 * D);

    // ---- 8) LN -> FFN (gelu) -> residual output ----
    ln_kernel<<<dim3(M), b256, 0, stream>>>(comb, ln_g, ln_b, lnbuf, D3);
    launch_gemm<ACT_GELU,64>(stream, lnbuf, w_f1, b_f1, h2,  M, 2 * D, D3,    2 * D, 0, nullptr, nullptr);
    launch_gemm<ACT_NONE,64>(stream, h2,    w_f2, b_f2, out, M, D,     2 * D, D,     0, nullptr, x);
}
